// LocalDIM_56229711839708
// MI455X (gfx1250) — matrix operand order repaired and checked by reading, not compiled, not run on hardware
//
#include <hip/hip_runtime.h>
#include <hip/hip_bf16.h>
#include <math.h>

#define EPSV 1e-5f
#define TEMPV 0.07f

typedef __attribute__((ext_vector_type(16))) __bf16 v16bf;
typedef __attribute__((ext_vector_type(8)))  __bf16 v8bf;
typedef __attribute__((ext_vector_type(8)))  float  v8f;

union BF16x16 { v16bf v; v8bf h[2]; };

__device__ __forceinline__ v8f wmma_bf16(v16bf a, v16bf b, v8f c) {
  return __builtin_amdgcn_wmma_f32_16x16x32_bf16(false, a, false, b, (short)0, c, false, false);
}

// Load a 16-element fragment from two 8-element (16B) chunks.
__device__ __forceinline__ v16bf ld_frag2(const __bf16* p0, const __bf16* p1) {
  BF16x16 u;
  u.h[0] = *(const v8bf*)p0;
  u.h[1] = *(const v8bf*)p1;
  return u.v;
}

// ---------------------------------------------------------------------------
// Kernel 0: convert weights to bf16. A1 = [lW1; lWs] (1024 x 1536), W2b (512x512)
// ---------------------------------------------------------------------------
__global__ void k_prep(const float* __restrict__ W1, const float* __restrict__ Ws,
                       const float* __restrict__ W2,
                       __bf16* __restrict__ A1, __bf16* __restrict__ W2b) {
  int idx = blockIdx.x * 256 + threadIdx.x;
  const int NA = 1024 * 1536;
  if (idx < NA) {
    int o = idx / 1536, c = idx - o * 1536;
    float v = (o < 512) ? W1[o * 1536 + c] : Ws[(o - 512) * 1536 + c];
    A1[idx] = (__bf16)v;
  } else {
    int k = idx - NA;
    if (k < 512 * 512) W2b[k] = (__bf16)W2[k];
  }
}

// ---------------------------------------------------------------------------
// Kernel 1: per-batch GEMM  Y(1024 x 8192) = A1(1024x1536) x X_b(1536x256)
// grid: 512 blocks (32 b * 2 ntiles * 8 mtiles), block 256 (8 waves)
// each wave: one 16-row M tile x 8 N tiles of 16.
// Double-buffered LDS staging (transpose + bf16 convert), prefetch 2 ahead.
// launch_bounds(256,1): allow full register allocation (no spills).
// ---------------------------------------------------------------------------
#define XP 40
__global__ void __launch_bounds__(256, 1)
k_gemm1(const float* __restrict__ x, const __bf16* __restrict__ A1,
        float* __restrict__ Y) {
  __shared__ __bf16 ldsXt[2][128 * XP];   // transposed tile: [t_local][k_local]
  int blk = blockIdx.x;
  int b  = blk >> 4;
  int nb = (blk >> 3) & 1;
  int mb = blk & 7;
  int t0 = nb * 128;
  int m0 = mb * 128;
  int tid = threadIdx.x;
  int lane = tid & 31, wave = tid >> 5;
  int lr = lane & 15, hi = lane >> 4;

  v8f acc[8];
#pragma unroll
  for (int j = 0; j < 8; j++)
#pragma unroll
    for (int r = 0; r < 8; r++) acc[j][r] = 0.f;

  const float* xb = x + (size_t)b * 1536 * 256 + t0;
  int krow = tid >> 3;          // 0..31
  int tcb  = (tid & 7) * 16;    // 0..112

  // stage one K-tile (32 x 128) transposed into LDS as bf16
  auto stage = [&](int kk, int buf) {
    const float4* s4 = (const float4*)(xb + (size_t)(kk + krow) * 256 + tcb);
    float4 f[4];
    f[0] = s4[0]; f[1] = s4[1]; f[2] = s4[2]; f[3] = s4[3];
    const float* fv = (const float*)f;
#pragma unroll
    for (int ii = 0; ii < 16; ii++)
      ldsXt[buf][(tcb + ii) * XP + krow] = (__bf16)fv[ii];
  };

  stage(0, 0);
  __syncthreads();

  int cur = 0;
  for (int kk = 0; kk < 1536; kk += 32) {
    if (kk + 96 <= 1536)   // prefetch 2 tiles ahead (in-bounds)
      __builtin_prefetch(xb + (size_t)(kk + 64 + krow) * 256 + tcb, 0, 1);
    if (kk + 32 < 1536)    // pipeline: stage next tile into the other buffer
      stage(kk + 32, cur ^ 1);

    // A fragment for this wave's 16 rows
    const __bf16* ar = A1 + (size_t)(m0 + wave * 16 + lr) * 1536 + kk + 8 * hi;
    v16bf af = ld_frag2(ar, ar + 16);

    // load all 8 B fragments, then issue the WMMA chain back-to-back
    v16bf bfr[8];
#pragma unroll
    for (int j = 0; j < 8; j++) {
      const __bf16* bp = &ldsXt[cur][(j * 16 + lr) * XP + 8 * hi];
      bfr[j] = ld_frag2(bp, bp + 16);
    }
    // keep all fragment loads ahead of the WMMA chain (distinct regs, one wait)
    __builtin_amdgcn_sched_barrier(0);
#pragma unroll
    for (int j = 0; j < 8; j++) acc[j] = wmma_bf16(af, bfr[j], acc[j]);

    __syncthreads();
    cur ^= 1;
  }

#pragma unroll
  for (int j = 0; j < 8; j++) {
    int col = b * 256 + t0 + j * 16 + lr;
#pragma unroll
    for (int r = 0; r < 8; r++) {
      int row = m0 + wave * 16 + r + 8 * hi;
      Y[(size_t)row * 8192 + col] = acc[j][r];
    }
  }
}

// ---------------------------------------------------------------------------
// Kernel 2: BatchNorm stats over (B,T) for channels 0..511 of Y
// ---------------------------------------------------------------------------
__global__ void k_bnstats(const float* __restrict__ Y, const float* __restrict__ g1,
                          const float* __restrict__ b1,
                          float* __restrict__ scale, float* __restrict__ shift) {
  int o = blockIdx.x;
  int tid = threadIdx.x;
  __shared__ float rs[256], rq[256];
  float s = 0.f, q = 0.f;
  const float* row = Y + (size_t)o * 8192;
  for (int i = tid; i < 8192; i += 256) { float v = row[i]; s += v; q += v * v; }
  rs[tid] = s; rq[tid] = q; __syncthreads();
  for (int st = 128; st > 0; st >>= 1) {
    if (tid < st) { rs[tid] += rs[tid + st]; rq[tid] += rq[tid + st]; }
    __syncthreads();
  }
  if (tid == 0) {
    float mu = rs[0] / 8192.f;
    float var = rq[0] / 8192.f - mu * mu;
    float sc = g1[o] * rsqrtf(var + EPSV);
    scale[o] = sc;
    shift[o] = b1[o] - mu * sc;
  }
}

// ---------------------------------------------------------------------------
// Kernel 3: global net (tiny): full mi_net on (32,192) -> normalized g (32x512)
// single block, 256 threads
// ---------------------------------------------------------------------------
__global__ void k_globalnet(const float* __restrict__ xg,
                            const float* __restrict__ gW1, const float* __restrict__ gg1,
                            const float* __restrict__ gb1, const float* __restrict__ gW2,
                            const float* __restrict__ gb2, const float* __restrict__ gWs,
                            const float* __restrict__ glng, const float* __restrict__ glnb,
                            float* __restrict__ gf, __bf16* __restrict__ gbf) {
  __shared__ float  xbuf[32 * 192];
  __shared__ float  hbuf[512 * 32];
  __shared__ __bf16 ybn[512 * 32];
  __shared__ float  scb[512], shb[512];
  __shared__ float  redA[8][32], redB[8][32];
  __shared__ float  muv[32], rsv[32], nrmv[32];
  int tid = threadIdx.x;

  for (int i = tid; i < 32 * 192; i += 256) xbuf[i] = xg[i];
  __syncthreads();

  // y1 = gW1 @ xg^T  -> hbuf[o][j]
  for (int idx = tid; idx < 512 * 32; idx += 256) {
    int o = idx >> 5, j = idx & 31;
    const float* wr = gW1 + o * 192;
    const float* xr = xbuf + j * 192;
    float a = 0.f;
    for (int c = 0; c < 192; c++) a += wr[c] * xr[c];
    hbuf[idx] = a;
  }
  __syncthreads();

  // BN over batch (32)
  for (int o = tid; o < 512; o += 256) {
    float s = 0.f, q = 0.f;
    for (int j = 0; j < 32; j++) { float v = hbuf[o * 32 + j]; s += v; q += v * v; }
    float mu = s / 32.f, var = q / 32.f - mu * mu;
    float sc = gg1[o] * rsqrtf(var + EPSV);
    scb[o] = sc; shb[o] = gb1[o] - mu * sc;
  }
  __syncthreads();
  for (int idx = tid; idx < 512 * 32; idx += 256) {
    int o = idx >> 5;
    ybn[idx] = (__bf16)fmaxf(hbuf[idx] * scb[o] + shb[o], 0.f);
  }
  __syncthreads();

  // h = gW2 @ ybn + gb2 + gWs @ xg^T
  for (int idx = tid; idx < 512 * 32; idx += 256) {
    int o = idx >> 5, j = idx & 31;
    float a = gb2[o];
    const float* w2 = gW2 + o * 512;
    for (int k = 0; k < 512; k++) a += w2[k] * (float)ybn[k * 32 + j];
    const float* wsr = gWs + o * 192;
    const float* xr = xbuf + j * 192;
    for (int c = 0; c < 192; c++) a += wsr[c] * xr[c];
    hbuf[idx] = a;
  }
  __syncthreads();

  // LayerNorm over 512 channels per sample j, then l2-normalize
  int seg = tid >> 5, j = tid & 31;
  float s = 0.f, q = 0.f;
  for (int r = 0; r < 64; r++) { float v = hbuf[(seg * 64 + r) * 32 + j]; s += v; q += v * v; }
  redA[seg][j] = s; redB[seg][j] = q; __syncthreads();
  if (tid < 32) {
    float ss = 0.f, qq = 0.f;
    for (int g = 0; g < 8; g++) { ss += redA[g][tid]; qq += redB[g][tid]; }
    float mu = ss / 512.f, var = qq / 512.f - mu * mu;
    muv[tid] = mu; rsv[tid] = rsqrtf(var + EPSV);
  }
  __syncthreads();
  float mu = muv[j], rs = rsv[j];
  float ssq = 0.f;
  for (int r = 0; r < 64; r++) {
    int o = seg * 64 + r;
    float v = (hbuf[o * 32 + j] - mu) * rs * glng[o] + glnb[o];
    ssq += v * v;
  }
  redA[seg][j] = ssq; __syncthreads();
  if (tid < 32) {
    float t = 0.f;
    for (int g = 0; g < 8; g++) t += redA[g][tid];
    nrmv[tid] = sqrtf(t);
  }
  __syncthreads();
  float inn = 1.f / nrmv[j];
  for (int r = 0; r < 64; r++) {
    int o = seg * 64 + r;
    float v = (hbuf[o * 32 + j] - mu) * rs * glng[o] + glnb[o];
    float gn = v * inn;
    gf[j * 512 + o]  = gn;
    gbf[j * 512 + o] = (__bf16)gn;
  }
}

// ---------------------------------------------------------------------------
// Kernel 4: GEMM2 (W2 x relu(bn(Y1))) + bias + shortcut + LayerNorm +
//           l2-normalize + positive logits u_p.  One block per 32 positions.
// grid 256 blocks, block 256 (8 waves; wave w owns rows [64w,64w+64))
// Double-buffered LDS staging with BN+relu fused into the stage.
// ---------------------------------------------------------------------------
#define XP2 40
__global__ void __launch_bounds__(256, 1)
k_gemm2_ln(const float* __restrict__ Y, const __bf16* __restrict__ W2b,
           const float* __restrict__ bnscale, const float* __restrict__ bnshift,
           const float* __restrict__ b2, const float* __restrict__ lng,
           const float* __restrict__ lnb, const float* __restrict__ gf,
           __bf16* __restrict__ Lp, float* __restrict__ up) {
  __shared__ __bf16 ldsBt[2][32 * XP2];  // transposed relu(BN(Y1)) tile [col][k]
  __shared__ float  hbuf[512 * 33];      // pre-LN activations
  __shared__ float  gbuf[512];
  __shared__ float  redA[8][32], redB[8][32];
  __shared__ float  muv[32], rsv[32], nrmv[32];

  int p0 = blockIdx.x * 32;
  int n  = p0 >> 8;                       // sample index (constant per block)
  int tid = threadIdx.x;
  int lane = tid & 31, wave = tid >> 5;
  int lr = lane & 15, hi = lane >> 4;

  for (int i = tid; i < 512; i += 256) gbuf[i] = gf[n * 512 + i];

  v8f acc[4][2];
#pragma unroll
  for (int i = 0; i < 4; i++)
#pragma unroll
    for (int j = 0; j < 2; j++)
#pragma unroll
      for (int r = 0; r < 8; r++) acc[i][j][r] = 0.f;

  int krow = tid >> 3;          // 0..31
  int cb   = (tid & 7) * 4;     // 0..28

  auto stage = [&](int kk, int buf) {
    int c = kk + krow;
    float4 f4 = *(const float4*)(Y + (size_t)c * 8192 + p0 + cb);
    float sc = bnscale[c], sh = bnshift[c];
    ldsBt[buf][(cb + 0) * XP2 + krow] = (__bf16)fmaxf(f4.x * sc + sh, 0.f);
    ldsBt[buf][(cb + 1) * XP2 + krow] = (__bf16)fmaxf(f4.y * sc + sh, 0.f);
    ldsBt[buf][(cb + 2) * XP2 + krow] = (__bf16)fmaxf(f4.z * sc + sh, 0.f);
    ldsBt[buf][(cb + 3) * XP2 + krow] = (__bf16)fmaxf(f4.w * sc + sh, 0.f);
  };

  stage(0, 0);
  __syncthreads();

  int cur = 0;
  for (int kk = 0; kk < 512; kk += 32) {
    if (kk + 32 < 512) stage(kk + 32, cur ^ 1);

    v16bf af[4];
#pragma unroll
    for (int i = 0; i < 4; i++) {
      const __bf16* ar = W2b + (size_t)(wave * 64 + i * 16 + lr) * 512 + kk + 8 * hi;
      af[i] = ld_frag2(ar, ar + 16);
    }
    v16bf bfr[2];
#pragma unroll
    for (int j = 0; j < 2; j++) {
      const __bf16* bp = &ldsBt[cur][(j * 16 + lr) * XP2 + 8 * hi];
      bfr[j] = ld_frag2(bp, bp + 16);
    }
    __builtin_amdgcn_sched_barrier(0);
#pragma unroll
    for (int j = 0; j < 2; j++)
#pragma unroll
      for (int i = 0; i < 4; i++) acc[i][j] = wmma_bf16(af[i], bfr[j], acc[i][j]);

    __syncthreads();
    cur ^= 1;
  }

  // h = acc + b2 + shortcut -> hbuf
#pragma unroll
  for (int i = 0; i < 4; i++)
#pragma unroll
    for (int j = 0; j < 2; j++) {
      int col = j * 16 + lr;
#pragma unroll
      for (int r = 0; r < 8; r++) {
        int row = wave * 64 + i * 16 + r + 8 * hi;
        float h = acc[i][j][r] + b2[row] + Y[(size_t)(512 + row) * 8192 + p0 + col];
        hbuf[row * 33 + col] = h;
      }
    }
  __syncthreads();

  // LayerNorm over 512 channels per position
  int seg = tid >> 5, col = tid & 31;
  float s = 0.f, q = 0.f;
  for (int r = 0; r < 64; r++) { float v = hbuf[(seg * 64 + r) * 33 + col]; s += v; q += v * v; }
  redA[seg][col] = s; redB[seg][col] = q; __syncthreads();
  if (tid < 32) {
    float ss = 0.f, qq = 0.f;
    for (int g = 0; g < 8; g++) { ss += redA[g][tid]; qq += redB[g][tid]; }
    float mu = ss / 512.f, var = qq / 512.f - mu * mu;
    muv[tid] = mu; rsv[tid] = rsqrtf(var + EPSV);
  }
  __syncthreads();
  float mu = muv[col], rs = rsv[col];
  float ssq = 0.f, dt = 0.f;
  for (int r = 0; r < 64; r++) {
    int row = seg * 64 + r;
    float o = (hbuf[row * 33 + col] - mu) * rs * lng[row] + lnb[row];
    ssq += o * o; dt += o * gbuf[row];
  }
  redA[seg][col] = ssq; redB[seg][col] = dt; __syncthreads();
  if (tid < 32) {
    float ss = 0.f, dd = 0.f;
    for (int g = 0; g < 8; g++) { ss += redA[g][tid]; dd += redB[g][tid]; }
    float nrm = sqrtf(ss);
    nrmv[tid] = nrm;
    up[p0 + tid] = dd / (nrm * TEMPV);
  }
  __syncthreads();
  float inn = 1.f / nrmv[col];
  for (int r = 0; r < 64; r++) {
    int row = seg * 64 + r;
    float o = (hbuf[row * 33 + col] - mu) * rs * lng[row] + lnb[row];
    Lp[(size_t)(p0 + col) * 512 + row] = (__bf16)(o * inn);
  }
}

// ---------------------------------------------------------------------------
// Kernel 5: negatives u_n = G(32x512) x Lp^T, exp + mask + per-n partial sums.
// grid 64 blocks x 8 waves; each wave one 16-column p tile, 2 M tiles.
// ---------------------------------------------------------------------------
__global__ void __launch_bounds__(256, 1)
k_neg(const __bf16* __restrict__ gbf, const __bf16* __restrict__ Lp,
      float* __restrict__ partial) {
  __shared__ float exsum[8][32];
  int tid = threadIdx.x;
  int lane = tid & 31, wave = tid >> 5;
  int lr = lane & 15, hi = lane >> 4;
  int p0 = (blockIdx.x * 8 + wave) * 16;

  v8f a0, a1;
#pragma unroll
  for (int r = 0; r < 8; r++) { a0[r] = 0.f; a1[r] = 0.f; }

  for (int kk = 0; kk < 512; kk += 32) {
    const __bf16* bp = Lp + (size_t)(p0 + lr) * 512 + kk + 8 * hi;
    v16bf bf = ld_frag2(bp, bp + 16);
    const __bf16* ar0 = gbf + (size_t)lr * 512 + kk + 8 * hi;
    const __bf16* ar1 = gbf + (size_t)(16 + lr) * 512 + kk + 8 * hi;
    v16bf af0 = ld_frag2(ar0, ar0 + 16);
    v16bf af1 = ld_frag2(ar1, ar1 + 16);
    __builtin_amdgcn_sched_barrier(0);
    a0 = wmma_bf16(af0, bf, a0);
    a1 = wmma_bf16(af1, bf, a1);
  }

  int p  = p0 + lr;
  int pn = p >> 8;    // sample owning this position (for masking)
#pragma unroll
  for (int r = 0; r < 8; r++) {
    int n0 = r + 8 * hi;
    float e0 = (pn == n0) ? 0.f : expf(a0[r]);
#pragma unroll
    for (int m = 1; m < 16; m <<= 1) e0 += __shfl_xor(e0, m, 16);
    if (lr == 0) exsum[wave][n0] = e0;
    int n1 = 16 + r + 8 * hi;
    float e1 = (pn == n1) ? 0.f : expf(a1[r]);
#pragma unroll
    for (int m = 1; m < 16; m <<= 1) e1 += __shfl_xor(e1, m, 16);
    if (lr == 0) exsum[wave][n1] = e1;
  }
  __syncthreads();
  if (tid < 32) {
    float s = 0.f;
    for (int w = 0; w < 8; w++) s += exsum[w][tid];
    partial[blockIdx.x * 32 + tid] = s;
  }
}

// ---------------------------------------------------------------------------
// Kernel 6: finish: A[n] = sum partials; loss = mean(log(e^{u_p}+A_n) - u_p)
// ---------------------------------------------------------------------------
__global__ void k_finish(const float* __restrict__ partial, const float* __restrict__ up,
                         float* __restrict__ out) {
  __shared__ float An[32];
  __shared__ float red[256];
  int tid = threadIdx.x;
  if (tid < 32) {
    float s = 0.f;
    for (int b = 0; b < 64; b++) s += partial[b * 32 + tid];
    An[tid] = s;
  }
  __syncthreads();
  float acc = 0.f;
  for (int p = tid; p < 8192; p += 256) {
    float u = up[p];
    acc += logf(expf(u) + An[p >> 8]) - u;
  }
  red[tid] = acc; __syncthreads();
  for (int st = 128; st > 0; st >>= 1) {
    if (tid < st) red[tid] += red[tid + st];
    __syncthreads();
  }
  if (tid == 0) out[0] = red[0] / 8192.f;
}

// ---------------------------------------------------------------------------
extern "C" void kernel_launch(void* const* d_in, const int* in_sizes, int n_in,
                              void* d_out, int out_size, void* d_ws, size_t ws_size,
                              hipStream_t stream) {
  (void)in_sizes; (void)n_in; (void)out_size; (void)ws_size;
  const float* x    = (const float*)d_in[0];
  const float* xg   = (const float*)d_in[1];
  const float* lW1  = (const float*)d_in[2];
  const float* lg1  = (const float*)d_in[3];
  const float* lb1  = (const float*)d_in[4];
  const float* lW2  = (const float*)d_in[5];
  const float* lb2  = (const float*)d_in[6];
  const float* lWs  = (const float*)d_in[7];
  const float* llng = (const float*)d_in[8];
  const float* llnb = (const float*)d_in[9];
  const float* gW1  = (const float*)d_in[10];
  const float* gg1  = (const float*)d_in[11];
  const float* gb1  = (const float*)d_in[12];
  const float* gW2  = (const float*)d_in[13];
  const float* gb2  = (const float*)d_in[14];
  const float* gWs  = (const float*)d_in[15];
  const float* glng = (const float*)d_in[16];
  const float* glnb = (const float*)d_in[17];

  char* w = (char*)d_ws;
  size_t off = 0;
  auto carve = [&](size_t bytes) -> char* {
    off = (off + 255) & ~(size_t)255;
    char* p = w + off;
    off += bytes;
    return p;
  };
  __bf16* A1      = (__bf16*)carve((size_t)1024 * 1536 * 2);
  __bf16* W2b     = (__bf16*)carve((size_t)512 * 512 * 2);
  float*  Y       = (float*) carve((size_t)1024 * 8192 * 4);
  float*  bnscale = (float*) carve(512 * 4);
  float*  bnshift = (float*) carve(512 * 4);
  float*  gf      = (float*) carve((size_t)32 * 512 * 4);
  __bf16* gbf     = (__bf16*)carve((size_t)32 * 512 * 2);
  __bf16* Lp      = (__bf16*)carve((size_t)8192 * 512 * 2);
  float*  up      = (float*) carve(8192 * 4);
  float*  partial = (float*) carve((size_t)64 * 32 * 4);

  k_prep<<<dim3((1024 * 1536 + 512 * 512 + 255) / 256), dim3(256), 0, stream>>>(
      lW1, lWs, lW2, A1, W2b);
  k_gemm1<<<dim3(512), dim3(256), 0, stream>>>(x, A1, Y);
  k_bnstats<<<dim3(512), dim3(256), 0, stream>>>(Y, lg1, lb1, bnscale, bnshift);
  k_globalnet<<<dim3(1), dim3(256), 0, stream>>>(
      xg, gW1, gg1, gb1, gW2, gb2, gWs, glng, glnb, gf, gbf);
  k_gemm2_ln<<<dim3(256), dim3(256), 0, stream>>>(
      Y, W2b, bnscale, bnshift, lb2, llng, llnb, gf, Lp, up);
  k_neg<<<dim3(64), dim3(256), 0, stream>>>(gbf, Lp, partial);
  k_finish<<<dim3(1), dim3(256), 0, stream>>>(partial, up, (float*)d_out);
}
